// CausalSelfAttention_78099685310517
// MI455X (gfx1250) — compile-verified, matrix operand order checked by reading, not run on hardware
//
#include <hip/hip_runtime.h>
#include <hip/hip_bf16.h>

typedef __attribute__((ext_vector_type(16))) _Float16 v16h;
typedef __attribute__((ext_vector_type(8)))  _Float16 v8h;
typedef __attribute__((ext_vector_type(4)))  _Float16 v4h;
typedef __attribute__((ext_vector_type(8)))  float    v8f;
typedef unsigned int u32;

#define WMMA_F16(a, b, c) \
  __builtin_amdgcn_wmma_f32_16x16x32_f16(false, (a), false, (b), (short)0, (c), false, false)

// ---- CDNA5 async global->LDS copy (ASYNCcnt path); parameter types are
// int4-vector pointers in AS1/AS3 (confirmed by round-2 diagnostics).
#if __has_builtin(__builtin_amdgcn_global_load_async_to_lds_b128)
#define HAVE_ASYNC 1
typedef int v4i __attribute__((__vector_size__(16)));
typedef __attribute__((address_space(1))) v4i* as1_v4i;
typedef __attribute__((address_space(3))) v4i* as3_v4i;
#define ASYNC_COPY16(gsrc, ldst)                                    \
  __builtin_amdgcn_global_load_async_to_lds_b128(                   \
      (as1_v4i)(__UINTPTR_TYPE__)(const void*)(gsrc),               \
      (as3_v4i)(__UINTPTR_TYPE__)(const void*)(ldst), 0, 0)
#if __has_builtin(__builtin_amdgcn_s_wait_asynccnt)
#define ASYNC_WAIT_N(n) __builtin_amdgcn_s_wait_asynccnt(n)
#else
#define ASYNC_WAIT_N(n) asm volatile("s_wait_asynccnt %0" ::"i"(n) : "memory")
#endif
#else
#define HAVE_ASYNC 0
#define ASYNC_WAIT_N(n)
#endif

constexpr int BB = 4, SS = 2048, DD = 1024, HH = 16, HD = 64;

// Gather one 16x32 f16 WMMA operand fragment from a row-major LDS tile.
// CDNA5 16-bit A/B layout (ISA 7.12.2): lanes 0-15 hold row M=lane with
// halves 0..7 -> K 0..7 and halves 8..15 -> K 16..23; lanes 16-31 same rows
// with +8 K offset.  Two 16B ds_load_b128 per fragment; ld padded (40/72)
// so every access is 16B aligned and bank-conflict free.
__device__ __forceinline__ v16h lds_frag(const _Float16* base, int row_base,
                                         int col_base, int ld) {
  const int lane = threadIdx.x & 31;
  const _Float16* p =
      base + (row_base + (lane & 15)) * ld + col_base + ((lane & 16) ? 8 : 0);
  v8h lo = *(const v8h*)p;
  v8h hi = *(const v8h*)(p + 16);
  return __builtin_shufflevector(lo, hi, 0, 1, 2, 3, 4, 5, 6, 7, 8, 9, 10, 11,
                                 12, 13, 14, 15);
}

__device__ __forceinline__ float redmax16(float x) {
#pragma unroll
  for (int m = 1; m < 16; m <<= 1) x = fmaxf(x, __shfl_xor(x, m, 32));
  return x;
}
__device__ __forceinline__ float redsum16(float x) {
#pragma unroll
  for (int m = 1; m < 16; m <<= 1) x += __shfl_xor(x, m, 32);
  return x;
}

// ---------------- Kernel 1: qkv = x @ w_qkv + b_qkv, scatter to Q/K/Vt f16 --
__global__ __launch_bounds__(256) void qkv_gemm(
    const float* __restrict__ x, const float* __restrict__ w,
    const float* __restrict__ bias, _Float16* __restrict__ qbuf,
    _Float16* __restrict__ kbuf, _Float16* __restrict__ vbuf) {
  __shared__ __align__(16) _Float16 At[128 * 40];
  __shared__ __align__(16) _Float16 Bt[64 * 40];
  __shared__ __align__(16) _Float16 Vs[64][136];  // V-tile transpose staging
  const int t = threadIdx.x, lane = t & 31, wid = t >> 5;
  const int wm = wid & 3, wn = wid >> 2;
  const int m0 = blockIdx.y * 128, n0 = blockIdx.x * 64;
  v8f c[2][2] = {};
  for (int kb2 = 0; kb2 < DD; kb2 += 32) {
    {  // stage A: x[m0..+128][kb2..+32] fp32 -> f16, row-major ld=40
      const int row = t >> 1, half = (t & 1) * 16;
      const float* src = x + (size_t)(m0 + row) * DD + kb2 + half;
      __builtin_prefetch(src + 32, 0, 0);  // next K tile (speculative)
#pragma unroll
      for (int j = 0; j < 4; ++j) {
        float4 f = *(const float4*)(src + j * 4);
        v4h h;
        h[0] = (_Float16)f.x; h[1] = (_Float16)f.y;
        h[2] = (_Float16)f.z; h[3] = (_Float16)f.w;
        *(v4h*)(&At[row * 40 + half + j * 4]) = h;
      }
    }
#pragma unroll
    for (int p = 0; p < 2; ++p) {  // stage B^T: w[kb2..+32][n0..+64] -> Bt[n][k]
      const int kk = (t >> 4) + p * 16, nq = (t & 15) * 4;
      const float* src = w + (size_t)(kb2 + kk) * (3 * DD) + n0 + nq;
      __builtin_prefetch(src + (size_t)32 * (3 * DD), 0, 0);
      float4 f = *(const float4*)src;
      Bt[(nq + 0) * 40 + kk] = (_Float16)f.x;
      Bt[(nq + 1) * 40 + kk] = (_Float16)f.y;
      Bt[(nq + 2) * 40 + kk] = (_Float16)f.z;
      Bt[(nq + 3) * 40 + kk] = (_Float16)f.w;
    }
    __syncthreads();
    v16h a0 = lds_frag(At, wm * 32, 0, 40);
    v16h a1 = lds_frag(At, wm * 32 + 16, 0, 40);
    v16h b0 = lds_frag(Bt, wn * 32, 0, 40);
    v16h b1 = lds_frag(Bt, wn * 32 + 16, 0, 40);
    c[0][0] = WMMA_F16(a0, b0, c[0][0]);
    c[0][1] = WMMA_F16(a0, b1, c[0][1]);
    c[1][0] = WMMA_F16(a1, b0, c[1][0]);
    c[1][1] = WMMA_F16(a1, b1, c[1][1]);
    __syncthreads();
  }
  // epilogue: q/k/v section and head index are BLOCK-UNIFORM (n0 is 64-aligned
  // and heads are 64 wide) -> scalar branch, no per-lane exec divergence.
  auto epi = [&](auto&& store) {
#pragma unroll
    for (int im = 0; im < 2; ++im)
#pragma unroll
      for (int in2 = 0; in2 < 2; ++in2) {
        const int gn = n0 + wn * 32 + in2 * 16 + (lane & 15);
        const float bv = bias[gn];
        const int hd = gn & 63;
#pragma unroll
        for (int e = 0; e < 8; ++e) {
          const int gm = m0 + wm * 32 + im * 16 + ((lane >> 4) << 3) + e;
          store(gm >> 11, gm & (SS - 1), hd, c[im][in2][e] + bv);
        }
      }
  };
  const int sect = n0 >> 10;               // uniform: which of q/k/v
  const int hh = (n0 & (DD - 1)) >> 6;     // uniform: head index
  if (sect == 0) {
    epi([&](int bbb, int srow, int hd, float v) {
      qbuf[((size_t)(bbb * HH + hh) * SS + srow) * HD + hd] =
          (_Float16)(v * 0.125f);
    });
  } else if (sect == 1) {
    epi([&](int bbb, int srow, int hd, float v) {
      kbuf[((size_t)(bbb * HH + hh) * SS + srow) * HD + hd] = (_Float16)v;
    });
  } else {
    // V is stored transposed [b,h,hd,S].  A direct scatter would hit 32
    // distinct cachelines per store; transpose through LDS instead and emit
    // contiguous 256B rows (uniform branch -> barrier is safe).
    epi([&](int, int srow, int hd, float v) {
      Vs[hd][srow & 127] = (_Float16)v;
    });
    __syncthreads();
    const int bbb = m0 >> 11, srow0 = m0 & (SS - 1);  // 128-row tile, one batch
    _Float16* dst = vbuf + ((size_t)(bbb * HH + hh) * HD) * SS + srow0;
    for (int i = t; i < 64 * 16; i += 256) {
      const int r = i >> 4, cc = i & 15;
      *(v8h*)(dst + (size_t)r * SS + cc * 8) = *(const v8h*)(&Vs[r][cc * 8]);
    }
  }
}

// ---------------- Kernel 2: causal flash attention, per (b,h, 128 queries) --
__global__ __launch_bounds__(256) void flash_attn(
    const _Float16* __restrict__ qbuf, const _Float16* __restrict__ kbuf,
    const _Float16* __restrict__ vbuf, _Float16* __restrict__ obuf) {
  __shared__ __align__(16) _Float16 Qt[128 * 72];
  __shared__ __align__(16) _Float16 Kt[2][32 * 72];  // double-buffered
  __shared__ __align__(16) _Float16 Vt[2][64 * 40];  // double-buffered
  __shared__ __align__(16) _Float16 Pt[8][16 * 40];
  const int t = threadIdx.x, lane = t & 31, wv = t >> 5;
  const int by = blockIdx.y;  // bb*H + h
  const int gq0 = blockIdx.x * 128;
  const size_t headoff = (size_t)by * SS * HD;

  // Issue one K tile (32x128B, 8 chunks/row) + one V tile (64x64B, 4 chunks/
  // row) into LDS buffer `buf`; 2 async ops per thread, completion in order.
  auto stage_kv = [&](int kb0, int buf) {
#if HAVE_ASYNC
    const _Float16* ks = kbuf + headoff + (size_t)kb0 * HD;
    const int r = t >> 3, cc = t & 7;
    ASYNC_COPY16(ks + (size_t)r * HD + cc * 8, &Kt[buf][r * 72 + cc * 8]);
    const _Float16* vsrc = vbuf + headoff + kb0;
    const int rv = t >> 2, cv = t & 3;
    ASYNC_COPY16(vsrc + (size_t)rv * SS + cv * 8, &Vt[buf][rv * 40 + cv * 8]);
#else
    const u32* ks = (const u32*)(kbuf + headoff + (size_t)kb0 * HD);
    u32* kd = (u32*)&Kt[buf][0];
    for (int i = t; i < 32 * 32; i += 256)
      kd[(i >> 5) * 36 + (i & 31)] = ks[(i >> 5) * 32 + (i & 31)];
    const u32* vs = (const u32*)(vbuf + headoff + kb0);
    u32* vd = (u32*)&Vt[buf][0];
    for (int i = t; i < 64 * 16; i += 256)
      vd[(i >> 4) * 20 + (i & 15)] = vs[(size_t)(i >> 4) * (SS / 2) + (i & 15)];
#endif
  };

  // stage the 128x64 Q tile (row = 128B = 8 x 16B chunks, ld=72 halves)
#if HAVE_ASYNC
  {
    const _Float16* qs = qbuf + headoff + (size_t)gq0 * HD;
#pragma unroll
    for (int j = 0; j < 4; ++j) {
      const int i = t + j * 256, r = i >> 3, cc = i & 7;
      ASYNC_COPY16(qs + (size_t)r * HD + cc * 8, &Qt[r * 72 + cc * 8]);
    }
  }
#else
  {
    const u32* qs = (const u32*)(qbuf + headoff + (size_t)gq0 * HD);
    u32* qd = (u32*)Qt;
    for (int i = t; i < 128 * 32; i += 256)
      qd[(i >> 5) * 36 + (i & 31)] = qs[(i >> 5) * 32 + (i & 31)];
  }
#endif
  stage_kv(0, 0);     // prefetch first K/V block behind the Q copies
  ASYNC_WAIT_N(2);    // in-order completion: Q is resident, buf0 may be in flight
  __syncthreads();
  const v16h qa0 = lds_frag(Qt, wv * 16, 0, 72);
  const v16h qa1 = lds_frag(Qt, wv * 16, 32, 72);
  float m_i[8], l_i[8];
  v8f o0 = {}, o1 = {}, o2 = {}, o3 = {};
#pragma unroll
  for (int e = 0; e < 8; ++e) { m_i[e] = -1e30f; l_i[e] = 0.f; }
  const int rbase = gq0 + wv * 16 + ((lane >> 4) << 3);
  const int nkb = (gq0 + 128) >> 5;  // causal: keys < gq0+128
  for (int kbi = 0; kbi < nkb; ++kbi) {
    const int buf = kbi & 1;
    if (kbi + 1 < nkb) {
      stage_kv((kbi + 1) * 32, buf ^ 1);  // prefetch next block
      ASYNC_WAIT_N(2);                    // current buffer resident
    } else {
      ASYNC_WAIT_N(0);
    }
    __syncthreads();
    const _Float16* kt = &Kt[buf][0];
    const _Float16* vt = &Vt[buf][0];
    const int kb0 = kbi * 32;
    v8f s0 = {}, s1 = {};
    s0 = WMMA_F16(qa0, lds_frag(kt, 0, 0, 72), s0);
    s1 = WMMA_F16(qa0, lds_frag(kt, 16, 0, 72), s1);
    s0 = WMMA_F16(qa1, lds_frag(kt, 0, 32, 72), s0);
    s1 = WMMA_F16(qa1, lds_frag(kt, 16, 32, 72), s1);
    const int kcol = kb0 + (lane & 15);
    _Float16* pw = &Pt[wv][0];
    const int prow = (lane >> 4) << 3;
#pragma unroll
    for (int e = 0; e < 8; ++e) {
      const int qrow = rbase + e;
      const float x0 = (kcol <= qrow) ? s0[e] : -1e30f;
      const float x1 = (kcol + 16 <= qrow) ? s1[e] : -1e30f;
      const float mt = redmax16(fmaxf(x0, x1));
      const float mnew = fmaxf(m_i[e], mt);
      const float alpha = __expf(m_i[e] - mnew);
      const float p0 = __expf(x0 - mnew);
      const float p1 = __expf(x1 - mnew);
      l_i[e] = l_i[e] * alpha + redsum16(p0 + p1);
      m_i[e] = mnew;
      o0[e] *= alpha; o1[e] *= alpha; o2[e] *= alpha; o3[e] *= alpha;
      pw[(prow + e) * 40 + (lane & 15)] = (_Float16)p0;        // per-wave LDS;
      pw[(prow + e) * 40 + 16 + (lane & 15)] = (_Float16)p1;   // DS in-order
    }
    const v16h pa = lds_frag(pw, 0, 0, 40);
    o0 = WMMA_F16(pa, lds_frag(vt, 0, 0, 40), o0);
    o1 = WMMA_F16(pa, lds_frag(vt, 16, 0, 40), o1);
    o2 = WMMA_F16(pa, lds_frag(vt, 32, 0, 40), o2);
    o3 = WMMA_F16(pa, lds_frag(vt, 48, 0, 40), o3);
    __syncthreads();  // done reading buf before next prefetch overwrites it
  }
  const int bb2 = by >> 4, hh = by & 15;
#pragma unroll
  for (int e = 0; e < 8; ++e) {
    const float inv = 1.0f / l_i[e];
    const int srow = gq0 + wv * 16 + ((lane >> 4) << 3) + e;
    const size_t base =
        (size_t)(bb2 * SS + srow) * DD + hh * HD + (lane & 15);
    obuf[base + 0]  = (_Float16)(o0[e] * inv);
    obuf[base + 16] = (_Float16)(o1[e] * inv);
    obuf[base + 32] = (_Float16)(o2[e] * inv);
    obuf[base + 48] = (_Float16)(o3[e] * inv);
  }
}

// ---------------- Kernel 3: out = O @ w_out + b_out (fp32 out) --------------
__global__ __launch_bounds__(256) void out_gemm(
    const _Float16* __restrict__ obuf, const float* __restrict__ w,
    const float* __restrict__ bias, float* __restrict__ out) {
  __shared__ __align__(16) _Float16 At[128 * 40];
  __shared__ __align__(16) _Float16 Bt[64 * 40];
  const int t = threadIdx.x, lane = t & 31, wid = t >> 5;
  const int wm = wid & 3, wn = wid >> 2;
  const int m0 = blockIdx.y * 128, n0 = blockIdx.x * 64;
  v8f c[2][2] = {};
  for (int kb2 = 0; kb2 < DD; kb2 += 32) {
    // A tile is already f16 in memory: rows of 64B, 4 x 16B chunks each
#if HAVE_ASYNC
    {
      const _Float16* os = obuf + (size_t)m0 * DD + kb2;
#pragma unroll
      for (int j = 0; j < 2; ++j) {
        const int i = t + j * 256, r = i >> 2, cc = i & 3;
        ASYNC_COPY16(os + (size_t)r * DD + cc * 8, &At[r * 40 + cc * 8]);
      }
    }
#else
    {
      const u32* os = (const u32*)(obuf + (size_t)m0 * DD + kb2);
      u32* ad = (u32*)At;
      for (int i = t; i < 128 * 16; i += 256)
        ad[(i >> 4) * 20 + (i & 15)] = os[(size_t)(i >> 4) * (DD / 2) + (i & 15)];
    }
#endif
#pragma unroll
    for (int p = 0; p < 2; ++p) {
      const int kk = (t >> 4) + p * 16, nq = (t & 15) * 4;
      const float* src = w + (size_t)(kb2 + kk) * DD + n0 + nq;
      __builtin_prefetch(src + (size_t)32 * DD, 0, 0);
      float4 f = *(const float4*)src;
      Bt[(nq + 0) * 40 + kk] = (_Float16)f.x;
      Bt[(nq + 1) * 40 + kk] = (_Float16)f.y;
      Bt[(nq + 2) * 40 + kk] = (_Float16)f.z;
      Bt[(nq + 3) * 40 + kk] = (_Float16)f.w;
    }
    ASYNC_WAIT_N(0);
    __syncthreads();
    v16h a0 = lds_frag(At, wm * 32, 0, 40);
    v16h a1 = lds_frag(At, wm * 32 + 16, 0, 40);
    v16h b0 = lds_frag(Bt, wn * 32, 0, 40);
    v16h b1 = lds_frag(Bt, wn * 32 + 16, 0, 40);
    c[0][0] = WMMA_F16(a0, b0, c[0][0]);
    c[0][1] = WMMA_F16(a0, b1, c[0][1]);
    c[1][0] = WMMA_F16(a1, b0, c[1][0]);
    c[1][1] = WMMA_F16(a1, b1, c[1][1]);
    __syncthreads();
  }
#pragma unroll
  for (int im = 0; im < 2; ++im)
#pragma unroll
    for (int in2 = 0; in2 < 2; ++in2) {
      const int gn = n0 + wn * 32 + in2 * 16 + (lane & 15);
      const float bv = bias[gn];
#pragma unroll
      for (int e = 0; e < 8; ++e) {
        const int gm = m0 + wm * 32 + im * 16 + ((lane >> 4) << 3) + e;
        out[(size_t)gm * DD + gn] = c[im][in2][e] + bv;
      }
    }
}

extern "C" void kernel_launch(void* const* d_in, const int* in_sizes, int n_in,
                              void* d_out, int out_size, void* d_ws,
                              size_t ws_size, hipStream_t stream) {
  (void)in_sizes; (void)n_in; (void)out_size; (void)ws_size;
  const float* x = (const float*)d_in[0];
  const float* w_qkv = (const float*)d_in[1];
  const float* b_qkv = (const float*)d_in[2];
  const float* w_out = (const float*)d_in[3];
  const float* b_out = (const float*)d_in[4];
  float* out = (float*)d_out;
  // workspace: Q,K (head-major), V^T, O  -- 4 x 16 MB f16 = 64 MB
  const size_t elems = (size_t)BB * HH * SS * HD;  // 8388608
  _Float16* qbuf = (_Float16*)d_ws;
  _Float16* kbuf = qbuf + elems;
  _Float16* vbuf = kbuf + elems;
  _Float16* obuf = vbuf + elems;

  qkv_gemm<<<dim3((3 * DD) / 64, (BB * SS) / 128), 256, 0, stream>>>(
      x, w_qkv, b_qkv, qbuf, kbuf, vbuf);
  flash_attn<<<dim3(SS / 128, BB * HH), 256, 0, stream>>>(qbuf, kbuf, vbuf,
                                                          obuf);
  out_gemm<<<dim3(DD / 64, (BB * SS) / 128), 256, 0, stream>>>(obuf, w_out,
                                                               b_out, out);
}
